// AdapterGNN_76330158785174
// MI455X (gfx1250) — matrix operand order repaired and checked by reading, not compiled, not run on hardware
//
#include <hip/hip_runtime.h>

#define IN_DIM  768
#define OUT_DIM 128
#define NNODES  100000
#define NEDGES  1600000

typedef __attribute__((ext_vector_type(16))) __bf16 v16bf;
typedef __attribute__((ext_vector_type(8)))  __bf16 v8bf;
typedef __attribute__((ext_vector_type(8)))  float  v8f;

// Convert 8 fp32 (two float4) into bf16 fragment slots [BASE..BASE+7], scaled.
template <int BASE>
__device__ __forceinline__ void cvt8(v16bf& a, const float4& x, const float4& y, float s) {
  a[BASE + 0] = (__bf16)(x.x * s); a[BASE + 1] = (__bf16)(x.y * s);
  a[BASE + 2] = (__bf16)(x.z * s); a[BASE + 3] = (__bf16)(x.w * s);
  a[BASE + 4] = (__bf16)(y.x * s); a[BASE + 5] = (__bf16)(y.y * s);
  a[BASE + 6] = (__bf16)(y.z * s); a[BASE + 7] = (__bf16)(y.w * s);
}

// ---------------------------------------------------------------------------
// Pack fp32 weight W[K][N] (row-major) into bf16 WMMA B-fragments.
// Tile (kt, nt) covers K rows [kt*32, kt*32+32), N cols [nt*16, nt*16+16).
// Within a tile (512 elems): lane L holds column n = nt*16 + (L&15),
// K values kt*32 + (L>>4)*8 + (j&7) + (j>>3)*16 for j=0..15 (ISA B striping).
// ---------------------------------------------------------------------------
__global__ void k_pack(const float* __restrict__ W, __bf16* __restrict__ P, int K, int N) {
  int idx = blockIdx.x * 256 + threadIdx.x;
  if (idx >= K * N) return;
  int t = idx >> 9;
  int r = idx & 511;
  int lane = r >> 4;
  int j = r & 15;
  int ntiles = N >> 4;
  int kt = t / ntiles;
  int nt = t - kt * ntiles;
  int k = kt * 32 + (lane >> 4) * 8 + (j & 7) + (j >> 3) * 16;
  int n = nt * 16 + (lane & 15);
  P[idx] = (__bf16)W[k * N + n];
}

__global__ void k_zero(float* __restrict__ p, int n4) {
  int i = blockIdx.x * 256 + threadIdx.x;
  if (i < n4) ((float4*)p)[i] = make_float4(0.f, 0.f, 0.f, 0.f);
}

__global__ void k_deg(const int* __restrict__ src, const int* __restrict__ dst,
                      float* __restrict__ cntOut, float* __restrict__ cntIn) {
  int e = blockIdx.x * 256 + threadIdx.x;
  if (e < NEDGES) {
    atomicAdd(&cntOut[src[e]], 1.0f);
    atomicAdd(&cntIn[dst[e]], 1.0f);
  }
}

__global__ void k_norm(const float* __restrict__ cntOut, const float* __restrict__ cntIn,
                       float* __restrict__ nS, float* __restrict__ nD) {
  int i = blockIdx.x * 256 + threadIdx.x;
  if (i < NNODES) {
    nS[i] = rsqrtf(fmaxf(cntOut[i], 1.0f));
    nD[i] = rsqrtf(fmaxf(cntIn[i], 1.0f));
  }
}

// ---------------------------------------------------------------------------
// GEMM1: h = (features @ Wd + bd) * norm_src[:,None]
// One block = 16 nodes; 8 waves each own one 16-col tile of the 128 outputs.
// K loop: 24 steps of v_wmma_f32_16x16x32_bf16 (fp32->bf16 convert on the fly).
// ---------------------------------------------------------------------------
__global__ __launch_bounds__(256) void k_gemm1(
    const float* __restrict__ feat, const float* __restrict__ bd,
    const __bf16* __restrict__ WdP, const float* __restrict__ normS,
    float* __restrict__ h) {
  const int lane = threadIdx.x & 31;
  const int wave = threadIdx.x >> 5;          // n-tile 0..7
  const int nodeBase = blockIdx.x * 16;
  const int rowA = nodeBase + (lane & 15);    // A row this lane owns
  const int kHalf = (lane >> 4) * 8;          // ISA A-layout half select
  const float* frow = feat + (size_t)rowA * IN_DIM;

  v8f c = {0.f, 0.f, 0.f, 0.f, 0.f, 0.f, 0.f, 0.f};
  for (int kt = 0; kt < IN_DIM / 32; ++kt) {
    const int k0 = kt * 32;
    float4 a0 = *(const float4*)(frow + k0 + kHalf);
    float4 a1 = *(const float4*)(frow + k0 + kHalf + 4);
    float4 a2 = *(const float4*)(frow + k0 + 16 + kHalf);
    float4 a3 = *(const float4*)(frow + k0 + 16 + kHalf + 4);
    v16bf a;
    cvt8<0>(a, a0, a1, 1.0f);
    cvt8<8>(a, a2, a3, 1.0f);
    const v16bf b = *(const v16bf*)(WdP + (size_t)(kt * 8 + wave) * 512 + lane * 16);
    c = __builtin_amdgcn_wmma_f32_16x16x32_bf16(false, a, false, b, (short)0, c, false, false);
  }

  const int n = wave * 16 + (lane & 15);
  const float bias = bd[n];
  const int rBase = nodeBase + ((lane >> 4) * 8);   // C rows: lanes<16 -> M=r, else M=8+r
  float4 ns0 = *(const float4*)(normS + rBase);
  float4 ns1 = *(const float4*)(normS + rBase + 4);
  float nsv[8] = {ns0.x, ns0.y, ns0.z, ns0.w, ns1.x, ns1.y, ns1.z, ns1.w};
#pragma unroll
  for (int r = 0; r < 8; ++r)
    h[(size_t)(rBase + r) * OUT_DIM + n] = (c[r] + bias) * nsv[r];
}

// ---------------------------------------------------------------------------
// Scatter: agg[dst] += h[src]  (h pre-scaled by norm_src).
// 32 lanes per edge, float4 gather (L2-resident) + 4 f32 atomics per lane.
// ---------------------------------------------------------------------------
__global__ __launch_bounds__(256) void k_scatter(
    const int* __restrict__ src, const int* __restrict__ dst,
    const float* __restrict__ h, float* __restrict__ agg) {
  int tid = blockIdx.x * 256 + threadIdx.x;
  int e = tid >> 5;
  int c = tid & 31;
  if (e >= NEDGES) return;
  int s = src[e];
  int d = dst[e];
  float4 v = ((const float4*)(h + (size_t)s * OUT_DIM))[c];
  float* a = agg + (size_t)d * OUT_DIM + c * 4;
  atomicAdd(a + 0, v.x);
  atomicAdd(a + 1, v.y);
  atomicAdd(a + 2, v.z);
  atomicAdd(a + 3, v.w);
}

// ---------------------------------------------------------------------------
// Fused GEMM2+GEMM3:
//   g   = (agg * norm_dst[:,None]) @ Wg + bg     (WMMA, K=128 -> 4 steps)
//   out = g @ Wu + bu                            (WMMA, 48 n-tiles, 6 per wave)
// g tile staged through LDS as bf16 to re-layout C-fragments into A-fragments.
// ---------------------------------------------------------------------------
__global__ __launch_bounds__(256) void k_gemm23(
    const float* __restrict__ agg, const float* __restrict__ normD,
    const __bf16* __restrict__ WgP, const float* __restrict__ bg,
    const __bf16* __restrict__ WuP, const float* __restrict__ bu,
    float* __restrict__ out) {
  __shared__ __align__(16) __bf16 gtile[16 * 128];

  const int lane = threadIdx.x & 31;
  const int wave = threadIdx.x >> 5;
  const int nodeBase = blockIdx.x * 16;
  const int m = lane & 15;
  const int kHalf = (lane >> 4) * 8;
  const int rowA = nodeBase + m;
  const float nd = normD[rowA];                 // fold D_in^-1/2 into A conversion
  const float* arow = agg + (size_t)rowA * OUT_DIM;

  // ---- GEMM2 ----
  v8f c = {0.f, 0.f, 0.f, 0.f, 0.f, 0.f, 0.f, 0.f};
#pragma unroll
  for (int kt = 0; kt < 4; ++kt) {
    const int k0 = kt * 32;
    float4 a0 = *(const float4*)(arow + k0 + kHalf);
    float4 a1 = *(const float4*)(arow + k0 + kHalf + 4);
    float4 a2 = *(const float4*)(arow + k0 + 16 + kHalf);
    float4 a3 = *(const float4*)(arow + k0 + 16 + kHalf + 4);
    v16bf a;
    cvt8<0>(a, a0, a1, nd);
    cvt8<8>(a, a2, a3, nd);
    const v16bf b = *(const v16bf*)(WgP + (size_t)(kt * 8 + wave) * 512 + lane * 16);
    c = __builtin_amdgcn_wmma_f32_16x16x32_bf16(false, a, false, b, (short)0, c, false, false);
  }

  // g tile -> LDS (bf16, row-major [16][128]); each wave owns 16 columns
  const int n = wave * 16 + m;
  const float bgv = bg[n];
  const int rOff = (lane >> 4) * 8;
#pragma unroll
  for (int r = 0; r < 8; ++r)
    gtile[(rOff + r) * 128 + n] = (__bf16)(c[r] + bgv);
  __syncthreads();

  // ---- GEMM3 ----
  v8f acc[6];
  const v8f vz = {0.f, 0.f, 0.f, 0.f, 0.f, 0.f, 0.f, 0.f};
#pragma unroll
  for (int i = 0; i < 6; ++i) acc[i] = vz;

#pragma unroll
  for (int kt = 0; kt < 4; ++kt) {
    const int k0 = kt * 32;
    v8bf g0 = *(const v8bf*)&gtile[m * 128 + k0 + kHalf];       // ds_load_b128
    v8bf g1 = *(const v8bf*)&gtile[m * 128 + k0 + 16 + kHalf];  // ds_load_b128
    v16bf a;
#pragma unroll
    for (int j = 0; j < 8; ++j) { a[j] = g0[j]; a[j + 8] = g1[j]; }
#pragma unroll
    for (int i = 0; i < 6; ++i) {
      const int nt = i * 8 + wave;
      const v16bf b = *(const v16bf*)(WuP + (size_t)(kt * 48 + nt) * 512 + lane * 16);
      acc[i] = __builtin_amdgcn_wmma_f32_16x16x32_bf16(false, a, false, b, (short)0, acc[i],
                                                       false, false);
    }
  }

#pragma unroll
  for (int i = 0; i < 6; ++i) {
    const int nt = i * 8 + wave;
    const int nn = nt * 16 + m;
    const float bias = bu[nn];
#pragma unroll
    for (int r = 0; r < 8; ++r)
      out[(size_t)(nodeBase + rOff + r) * IN_DIM + nn] = acc[i][r] + bias;
  }
}

// ---------------------------------------------------------------------------
extern "C" void kernel_launch(void* const* d_in, const int* in_sizes, int n_in,
                              void* d_out, int out_size, void* d_ws, size_t ws_size,
                              hipStream_t stream) {
  const float* feat = (const float*)d_in[0];
  const float* Wd   = (const float*)d_in[1];
  const float* bd   = (const float*)d_in[2];
  const float* Wg   = (const float*)d_in[3];
  const float* bg   = (const float*)d_in[4];
  const float* Wu   = (const float*)d_in[5];
  const float* bu   = (const float*)d_in[6];
  const int*   src  = (const int*)d_in[7];
  const int*   dst  = (const int*)d_in[8];
  float* out = (float*)d_out;

  // Workspace layout (all offsets 128B-aligned); agg+counters contiguous for one zero pass.
  char* ws = (char*)d_ws;
  float*  agg    = (float*)(ws + 0);          // 100000*128 f32 = 51,200,000 B
  float*  cntOut = (float*)(ws + 51200000);   //   400,000 B
  float*  cntIn  = (float*)(ws + 51600000);   //   400,000 B
  float*  normS  = (float*)(ws + 52000000);   //   400,000 B
  float*  normD  = (float*)(ws + 52400000);   //   400,000 B
  __bf16* WdP    = (__bf16*)(ws + 52800000);  //   196,608 B
  __bf16* WgP    = (__bf16*)(ws + 52996608);  //    32,768 B
  __bf16* WuP    = (__bf16*)(ws + 53029376);  //   196,608 B  (end ~53.2 MB)

  // h [100000,128] parked in d_out's memory (51 MB of 307 MB); consumed by the
  // scatter before k_gemm23 overwrites every element of d_out.
  float* h = out;

  k_pack<<<(IN_DIM * OUT_DIM + 255) / 256, 256, 0, stream>>>(Wd, WdP, IN_DIM, OUT_DIM);
  k_pack<<<(OUT_DIM * OUT_DIM + 255) / 256, 256, 0, stream>>>(Wg, WgP, OUT_DIM, OUT_DIM);
  k_pack<<<(OUT_DIM * IN_DIM + 255) / 256, 256, 0, stream>>>(Wu, WuP, OUT_DIM, IN_DIM);

  // zero agg + cntOut + cntIn = 52,000,000 B = 13,000,000 floats = 3,250,000 float4
  k_zero<<<(3250000 + 255) / 256, 256, 0, stream>>>((float*)ws, 3250000);

  k_deg<<<(NEDGES + 255) / 256, 256, 0, stream>>>(src, dst, cntOut, cntIn);
  k_norm<<<(NNODES + 255) / 256, 256, 0, stream>>>(cntOut, cntIn, normS, normD);

  k_gemm1<<<NNODES / 16, 256, 0, stream>>>(feat, bd, WdP, normS, h);
  k_scatter<<<(NEDGES * 32) / 256, 256, 0, stream>>>(src, dst, h, agg);
  k_gemm23<<<NNODES / 16, 256, 0, stream>>>(agg, normD, WgP, bg, WuP, bu, out);
}
